// LSTMDecoder_37778532336424
// MI455X (gfx1250) — matrix operand order repaired and checked by reading, not compiled, not run on hardware
//
#include <hip/hip_runtime.h>
#include <hip/hip_bf16.h>

// ---------------------------------------------------------------------------
// LSTM decoder for MI455X (gfx1250, wave32, WMMA).
// B=64 batch, L=64 latent, H=512 hidden, O=128 out, T=256 steps.
//
// Strategy:
//  * bf16 WMMA (v_wmma_f32_16x16x32_bf16) for all GEMMs, fp32 accumulate.
//  * 32 persistent workgroups, each owns 16 hidden units (one 16-wide N tile
//    per gate). Its 192KB bf16 weight slice is DMA'd into LDS ONCE with the
//    Tensor Data Mover and re-read every step with ds loads.
//  * h0/h1 state is exchanged through a double-buffered global staging area
//    (64KB bf16 each, L2-resident) pre-swizzled into the WMMA A layout.
//  * grid-wide sync = release/acquire atomic counter spin (2 per step).
//  * final projection = separate massively-parallel WMMA GEMM + sigmoid.
//
// Workspace layout (requires ws_size >= ~23.5 MB).
// ---------------------------------------------------------------------------

typedef __attribute__((ext_vector_type(16))) __bf16 v16bf;
typedef __attribute__((ext_vector_type(8)))  float  v8f;
typedef __attribute__((ext_vector_type(4)))  unsigned int u32x4;
typedef __attribute__((ext_vector_type(8)))  int    i32x8;
typedef __attribute__((ext_vector_type(4)))  int    i32x4;

#define NWG      32
#define WG_BYTES 196608ull   // per-WG packed weights: 64KB (layer0) + 128KB (layer1)

#define WPACK_OFF  0ull
#define WOUT_OFF   6291456ull     // 128KB packed W_out
#define BIAS0_OFF  6422528ull     // 2048 f32
#define BIAS1_OFF  6430720ull     // 2048 f32
#define H0S_OFF    6438912ull     // 2 x 64KB bf16 (double buffered, A-layout)
#define H1S_OFF    6569984ull     // 2 x 64KB
#define HS_OFF     6701056ull     // 16384x512 bf16 A-layout (16MB)
#define SYNC_OFF   23478272ull    // barrier counter

// ---- helpers ---------------------------------------------------------------

__device__ __forceinline__ unsigned short f2bf(float f) {
  unsigned u = __float_as_uint(f);
  unsigned r = u + 0x7FFFu + ((u >> 16) & 1u);   // round-to-nearest-even
  return (unsigned short)(r >> 16);
}

__device__ __forceinline__ float sigf(float x)  { return 1.0f / (1.0f + __expf(-x)); }
__device__ __forceinline__ float tanhf_f(float x){ float e = __expf(2.0f * x); return (e - 1.0f) / (e + 1.0f); }

__device__ __forceinline__ v8f splat8(float x) {
  v8f r;
#pragma unroll
  for (int i = 0; i < 8; ++i) r[i] = x;
  return r;
}

// WMMA A-matrix layout (16-bit, 16Mx32K per tile, ISA 7.12.2):
//   lane = (row%16) + 16*((k%32)>>3 & 1), vgpr v = ((k>>1)&3) | ((k>>4)<<2),
//   within-lane halfword = k&1.  Tile order: kTile-major, then mTile.
__device__ __forceinline__ unsigned a_off(int row, int k, int numMtiles) {
  int mt = row >> 4, r = row & 15;
  int kt = k >> 5,  kk = k & 31;
  int lane = r | (((kk >> 3) & 1) << 4);
  int v = ((kk >> 1) & 3) | ((kk >> 4) << 2);
  return (unsigned)(((kt * numMtiles + mt) << 10) + (lane << 5) + (v << 2) + ((kk & 1) << 1));
}

// dense-B layout (32Kx16N), same k order as A: lane = n + 16*((kk>>3)&1),
// vgpr v = ((kk>>1)&3) | ((kk>>4)<<2), 2 consecutive K per VGPR.
__device__ __forceinline__ unsigned b_off_in_tile(int n, int kk) {
  int lane = n + (((kk >> 3) & 1) << 4);
  int v = ((kk >> 1) & 3) | ((kk >> 4) << 2);
  return (unsigned)((lane << 5) + (v << 2) + ((kk & 1) << 1));
}

__device__ __forceinline__ v16bf ldmat(const unsigned char* p) {
  union { uint4 u[2]; v16bf v; } t;
  const uint4* q = (const uint4*)p;
  t.u[0] = q[0]; t.u[1] = q[1];
  return t.v;
}

__device__ __forceinline__ v8f wmma_bf16(v16bf a, v16bf b, v8f c) {
  return __builtin_amdgcn_wmma_f32_16x16x32_bf16(false, a, false, b, (short)0, c, false, false);
}

// release/acquire grid barrier over a monotonically increasing counter.
__device__ __forceinline__ void grid_sync(unsigned* ctr, unsigned target) {
  __threadfence();
  __syncthreads();
  if (threadIdx.x == 0) {
    __hip_atomic_fetch_add(ctr, 1u, __ATOMIC_ACQ_REL, __HIP_MEMORY_SCOPE_AGENT);
    while (__hip_atomic_load(ctr, __ATOMIC_ACQUIRE, __HIP_MEMORY_SCOPE_AGENT) < target)
      __builtin_amdgcn_s_sleep(2);
  }
  __syncthreads();
}

// ---- prep kernels ----------------------------------------------------------

// Pack a [2048, 512] gate-weight matrix (row = gate index) into per-WG
// B-operand tiles: ws[s*WG_BYTES + blkoff + (q*ktiles + kt)*1024 + intra].
__global__ __launch_bounds__(256) void pack_gate_w(const float* __restrict__ W,
                                                   unsigned char* __restrict__ dst,
                                                   int ktiles, int ktadd, int blkoff) {
  int e = blockIdx.x * blockDim.x + threadIdx.x;
  if (e >= 2048 * 512) return;
  int gi = e >> 9, k = e & 511;
  int q = gi >> 9, col = gi & 511;
  int s = col >> 4, n = col & 15;
  int kt = (k >> 5) + ktadd, kk = k & 31;
  unsigned long long off = (unsigned long long)s * WG_BYTES + (unsigned)blkoff +
                           (unsigned)(((q * ktiles + kt) << 10) + b_off_in_tile(n, kk));
  *(unsigned short*)(dst + off) = f2bf(W[e]);
}

// Pack W_out [128, 512] into B tiles: [n_t 0..7][k_t 0..15].
__global__ __launch_bounds__(256) void pack_wout(const float* __restrict__ W,
                                                 unsigned char* __restrict__ dst) {
  int e = blockIdx.x * blockDim.x + threadIdx.x;
  if (e >= 128 * 512) return;
  int o = e >> 9, k = e & 511;
  int nt = o >> 4, n = o & 15;
  int kt = k >> 5, kk = k & 31;
  *(unsigned short*)(dst + (((nt * 16 + kt) << 10) + b_off_in_tile(n, kk))) = f2bf(W[e]);
}

__global__ __launch_bounds__(256) void bias_combine(const float* b_ih0, const float* b_hh0,
                                                    const float* b_ih1, const float* b_hh1,
                                                    float* bias0, float* bias1) {
  int i = blockIdx.x * blockDim.x + threadIdx.x;
  if (i >= 2048) return;
  bias0[i] = b_ih0[i] + b_hh0[i];
  bias1[i] = b_ih1[i] + b_hh1[i];
}

// h_init = z @ W_init^T + b_init, written in bf16 A-layout to both h0/h1 buf0.
__global__ __launch_bounds__(256) void h_init_kernel(const float* __restrict__ z,
                                                     const float* __restrict__ Wi,
                                                     const float* __restrict__ bi,
                                                     unsigned char* h0s, unsigned char* h1s) {
  int tid = blockIdx.x * blockDim.x + threadIdx.x;
  if (tid >= 64 * 512) return;
  int b = tid >> 9, j = tid & 511;
  float acc = bi[j];
#pragma unroll 8
  for (int l = 0; l < 64; ++l) acc += z[b * 64 + l] * Wi[j * 64 + l];
  unsigned short hv = f2bf(acc);
  unsigned off = a_off(b, j, 4);
  *(unsigned short*)(h0s + off) = hv;
  *(unsigned short*)(h1s + off) = hv;
}

// ---- persistent LSTM kernel ------------------------------------------------

__global__ __launch_bounds__(256, 1) void lstm_persistent(
    const unsigned char* __restrict__ wpack,
    const float* __restrict__ bias0, const float* __restrict__ bias1,
    unsigned char* h0s, unsigned char* h1s, unsigned char* hs,
    unsigned* syncc) {
  __shared__ __align__(16) unsigned char wl[196608];  // packed weight slice
  __shared__ float gbuf[4096];                        // [gate q][b*16 + j]

  const int s = blockIdx.x;                           // hidden slice [16s, 16s+16)
  const int tid = threadIdx.x, lane = tid & 31, wv = tid >> 5;
  const int q = wv >> 1;                              // gate tile: 0=i 1=f 2=g 3=o
  const int mb = (wv & 1) * 2;                        // this wave's two M tiles

  // ---- one-shot weight DMA into LDS via the Tensor Data Mover (1D, 8B units)
  if (tid == 0) {
    unsigned lds = (unsigned)(size_t)(&wl[0]);
    unsigned long long ga = (unsigned long long)(size_t)(wpack + (unsigned long long)s * WG_BYTES);
    unsigned n8 = (unsigned)(WG_BYTES / 8ull);        // 24576 (fits tile_dim0)
    u32x4 g0; i32x8 g1; i32x4 g2; i32x4 g3; i32x8 g4;
    g0[0] = 1u;                                        // count=1, user descriptor
    g0[1] = lds;                                       // lds_addr
    g0[2] = (unsigned)(ga & 0xFFFFFFFFull);            // global_addr[31:0]
    g0[3] = (unsigned)((ga >> 32) & 0x1FFFFFFull) | (2u << 30);  // addr hi | type=2
    g1[0] = (3 << 16);                                 // data_size = 8B units
    g1[1] = (int)((n8 & 0xFFFFu) << 16);               // tensor_dim0[15:0]
    g1[2] = (int)((n8 >> 16) & 0xFFFFu) | (1 << 16);   // tensor_dim0[31:16], tensor_dim1=1
    g1[3] = (int)((n8 & 0xFFFFu) << 16);               // tile_dim0 = n8
    g1[4] = 0;                                         // tile_dim1/2 unused
    g1[5] = (int)n8;                                   // tensor_dim0_stride
    g1[6] = 0; g1[7] = 0;
    g2[0] = g2[1] = g2[2] = g2[3] = 0;
    g3[0] = g3[1] = g3[2] = g3[3] = 0;
    g4[0] = g4[1] = g4[2] = g4[3] = 0;
    g4[4] = g4[5] = g4[6] = g4[7] = 0;
    __builtin_amdgcn_tensor_load_to_lds(g0, g1, g2, g3, g4, 0);
    __builtin_amdgcn_s_wait_tensorcnt(0);
  }
  __syncthreads();

  const float bv0 = bias0[q * 512 + s * 16 + (lane & 15)];
  const float bv1 = bias1[q * 512 + s * 16 + (lane & 15)];
  float c0v[4] = {0.f, 0.f, 0.f, 0.f};
  float c1v[4] = {0.f, 0.f, 0.f, 0.f};
  unsigned phase = 0;

  for (int t = 0; t < 256; ++t) {
    const int rb = t & 1, wb = rb ^ 1;                 // double buffers
    const unsigned char* h0r = h0s + rb * 65536;
    unsigned char*       h0w = h0s + wb * 65536;
    const unsigned char* h1r = h1s + rb * 65536;
    unsigned char*       h1w = h1s + wb * 65536;

    // ---- layer 0: gates0 = bias0 + h0 @ W_hh0^T  (K = 512)
    {
      v8f a0 = splat8(bv0), a1 = splat8(bv0);
#pragma unroll 4
      for (int kt = 0; kt < 16; ++kt) {
        v16bf bm = ldmat(&wl[((q * 16 + kt) << 10) + (lane << 5)]);
        v16bf x0 = ldmat(h0r + ((kt * 4 + mb) << 10) + (lane << 5));
        v16bf x1 = ldmat(h0r + ((kt * 4 + mb + 1) << 10) + (lane << 5));
        a0 = wmma_bf16(x0, bm, a0);
        a1 = wmma_bf16(x1, bm, a1);
      }
      // C/D layout: lane%16 = N col, vgpr r = M row r (lanes<16) or r+8
      const int j = lane & 15, radd = (lane >> 4) << 3;
#pragma unroll
      for (int r = 0; r < 8; ++r) {
        gbuf[q * 1024 + ((mb * 16 + r + radd) << 4) + j]       = a0[r];
        gbuf[q * 1024 + (((mb + 1) * 16 + r + radd) << 4) + j] = a1[r];
      }
    }
    __syncthreads();
    // cell 0 update: 4 (b, j) pairs per thread
#pragma unroll
    for (int p = 0; p < 4; ++p) {
      int idx = tid * 4 + p, b = idx >> 4, j = idx & 15;
      float gi = sigf(gbuf[idx]);
      float gf = sigf(gbuf[1024 + idx]);
      float gg = tanhf_f(gbuf[2048 + idx]);
      float go = sigf(gbuf[3072 + idx]);
      c0v[p] = gf * c0v[p] + gi * gg;
      float h = go * tanhf_f(c0v[p]);
      *(unsigned short*)(h0w + a_off(b, s * 16 + j, 4)) = f2bf(h);
    }
    grid_sync(syncc, ++phase * NWG);                   // publish new h0

    // ---- layer 1: gates1 = bias1 + h0_new @ W_ih1^T + h1 @ W_hh1^T (K=1024)
    {
      v8f a0 = splat8(bv1), a1 = splat8(bv1);
#pragma unroll 4
      for (int kt = 0; kt < 32; ++kt) {
        v16bf bm = ldmat(&wl[65536 + ((q * 32 + kt) << 10) + (lane << 5)]);
        const unsigned char* ap = (kt < 16) ? (h0w + (kt << 12)) : (h1r + ((kt - 16) << 12));
        v16bf x0 = ldmat(ap + (mb << 10) + (lane << 5));
        v16bf x1 = ldmat(ap + ((mb + 1) << 10) + (lane << 5));
        a0 = wmma_bf16(x0, bm, a0);
        a1 = wmma_bf16(x1, bm, a1);
      }
      const int j = lane & 15, radd = (lane >> 4) << 3;
#pragma unroll
      for (int r = 0; r < 8; ++r) {
        gbuf[q * 1024 + ((mb * 16 + r + radd) << 4) + j]       = a0[r];
        gbuf[q * 1024 + (((mb + 1) * 16 + r + radd) << 4) + j] = a1[r];
      }
    }
    __syncthreads();
    // cell 1 update + emit h1 into hs (A-layout, row = t*64 + b, 1024 M tiles)
#pragma unroll
    for (int p = 0; p < 4; ++p) {
      int idx = tid * 4 + p, b = idx >> 4, j = idx & 15;
      float gi = sigf(gbuf[idx]);
      float gf = sigf(gbuf[1024 + idx]);
      float gg = tanhf_f(gbuf[2048 + idx]);
      float go = sigf(gbuf[3072 + idx]);
      c1v[p] = gf * c1v[p] + gi * gg;
      float h = go * tanhf_f(c1v[p]);
      unsigned short hb = f2bf(h);
      *(unsigned short*)(h1w + a_off(b, s * 16 + j, 4)) = hb;
      *(unsigned short*)(hs + a_off(t * 64 + b, s * 16 + j, 1024)) = hb;
    }
    grid_sync(syncc, ++phase * NWG);                   // publish new h1
  }
}

// ---- output projection: sigmoid(hs @ W_out^T + b_out) ----------------------

__global__ __launch_bounds__(256) void out_gemm(const unsigned char* __restrict__ hs,
                                                const unsigned char* __restrict__ wout,
                                                const float* __restrict__ b_out,
                                                float* __restrict__ out) {
  const int m = blockIdx.x;                // M tile (16 rows of the 16384)
  const int lane = threadIdx.x & 31, nt = threadIdx.x >> 5;   // 8 N tiles
  v8f acc = splat8(b_out[nt * 16 + (lane & 15)]);
#pragma unroll 4
  for (int kt = 0; kt < 16; ++kt) {
    v16bf a = ldmat(hs + ((unsigned)(kt * 1024 + m) << 10) + (lane << 5));
    v16bf b = ldmat(wout + ((nt * 16 + kt) << 10) + (lane << 5));
    acc = wmma_bf16(a, b, acc);
  }
  const int o = nt * 16 + (lane & 15), radd = (lane >> 4) << 3;
#pragma unroll
  for (int r = 0; r < 8; ++r) {
    int row = m * 16 + r + radd;           // row = t*64 + b
    int b = row & 63, t = row >> 6;
    out[((b << 8) + t) * 128 + o] = sigf(acc[r]);
  }
}

// ---- host glue -------------------------------------------------------------

extern "C" void kernel_launch(void* const* d_in, const int* in_sizes, int n_in,
                              void* d_out, int out_size, void* d_ws, size_t ws_size,
                              hipStream_t stream) {
  (void)in_sizes; (void)n_in; (void)out_size; (void)ws_size;  // ws_size >= ~23.5MB assumed
  const float* z      = (const float*)d_in[0];
  const float* W_init = (const float*)d_in[1];
  const float* b_init = (const float*)d_in[2];
  // d_in[3] = W_ih0: mathematically unused (LSTM input x == 0)
  const float* W_hh0  = (const float*)d_in[4];
  const float* b_ih0  = (const float*)d_in[5];
  const float* b_hh0  = (const float*)d_in[6];
  const float* W_ih1  = (const float*)d_in[7];
  const float* W_hh1  = (const float*)d_in[8];
  const float* b_ih1  = (const float*)d_in[9];
  const float* b_hh1  = (const float*)d_in[10];
  const float* W_out  = (const float*)d_in[11];
  const float* b_out  = (const float*)d_in[12];

  unsigned char* ws = (unsigned char*)d_ws;
  unsigned char* wpack = ws + WPACK_OFF;
  unsigned char* wout  = ws + WOUT_OFF;
  float* bias0 = (float*)(ws + BIAS0_OFF);
  float* bias1 = (float*)(ws + BIAS1_OFF);
  unsigned char* h0s = ws + H0S_OFF;
  unsigned char* h1s = ws + H1S_OFF;
  unsigned char* hs  = ws + HS_OFF;
  unsigned* syncc = (unsigned*)(ws + SYNC_OFF);

  (void)hipMemsetAsync(syncc, 0, 256, stream);         // reset grid barrier

  pack_gate_w<<<4096, 256, 0, stream>>>(W_hh0, wpack, 16, 0, 0);
  pack_gate_w<<<4096, 256, 0, stream>>>(W_ih1, wpack, 32, 0, 65536);
  pack_gate_w<<<4096, 256, 0, stream>>>(W_hh1, wpack, 32, 16, 65536);
  pack_wout<<<256, 256, 0, stream>>>(W_out, wout);
  bias_combine<<<8, 256, 0, stream>>>(b_ih0, b_hh0, b_ih1, b_hh1, bias0, bias1);
  h_init_kernel<<<128, 256, 0, stream>>>(z, W_init, b_init, h0s, h1s);

  lstm_persistent<<<NWG, 256, 0, stream>>>(wpack, bias0, bias1, h0s, h1s, hs, syncc);

  out_gemm<<<1024, 256, 0, stream>>>(hs, wout, b_out, (float*)d_out);
}
